// OuterProduct_15796889715495
// MI455X (gfx1250) — matrix operand order repaired and checked by reading, not compiled, not run on hardware
//
#include <hip/hip_runtime.h>
#include <hip/hip_bf16.h>

typedef __attribute__((ext_vector_type(16))) _Float16 v16h;
typedef __attribute__((ext_vector_type(8)))  _Float16 v8h;
typedef __attribute__((ext_vector_type(8)))  float    v8f;
typedef __attribute__((ext_vector_type(4)))  unsigned int v4u;
typedef __attribute__((ext_vector_type(8)))  int v8i;
typedef __attribute__((ext_vector_type(4)))  int v4i;

#define B_TOT 16384
#define F_TOT 39
#define E_TOT 64
#define U_TOT 128
#define K_TOT 4096   /* E*E */
#define NT    8      /* U / 16 */
#define KCN   128    /* K_TOT / 32 */
#define KB    16     /* kc-chunks per LDS staging window (16KB) */
#define WT    2      /* batch tiles (of 16) per wave */
#define WAVES 4      /* waves per block (wave32) */
#define GRID_Y 2
#define NT_PER (NT / GRID_Y)
#define NWIN  (NT_PER * (KCN / KB))

union V16 { v16h v; v8h h[2]; };

// ---------------------------------------------------------------------------
// TDM: 1-D contiguous copy global -> LDS via Tensor Data Mover (D# per ISA
// CDNA5 ISA). nelem8 = number of 8-byte elements.
// 6-arg builtin variant (clang-23 / therock-10.0 toolchain).
// ---------------------------------------------------------------------------
__device__ __forceinline__ void tdm_load_1d(unsigned lds_off, const void* gsrc,
                                            unsigned nelem8) {
  const unsigned long long ga = (unsigned long long)(uintptr_t)gsrc;
  v4u g0;
  g0.x = 1u;                                        // count=1, user load
  g0.y = lds_off;                                   // lds_addr (bytes)
  g0.z = (unsigned)(ga & 0xFFFFFFFFu);              // global_addr[31:0]
  g0.w = (unsigned)((ga >> 32) & 0x01FFFFFFu)       // global_addr[56:32]
       | (2u << 30);                                // type = 2 ("image")
  v8i g1;
  g1[0] = (3 << 16);                                // wg_mask=0, data_size=8B
  g1[1] = (int)((nelem8 & 0xFFFFu) << 16);          // tensor_dim0[15:0]
  g1[2] = (int)(((nelem8 >> 16) & 0xFFFFu)          // tensor_dim0[31:16]
       | (1u << 16));                               // tensor_dim1 = 1 (lo)
  g1[3] = (int)((nelem8 & 0xFFFFu) << 16);          // tensor_dim1 hi=0 | tile_dim0
  g1[4] = 1;                                        // tile_dim1=1, tile_dim2=0
  g1[5] = (int)nelem8;                              // tensor_dim0_stride[31:0]
  g1[6] = 0;                                        // stride0 hi | stride1 lo
  g1[7] = 0;                                        // stride1 hi
  const v4i z4 = {};
  const v8i z8 = {};
  __builtin_amdgcn_tensor_load_to_lds(g0, g1, z4, z4, z8, 0);
}

// ---------------------------------------------------------------------------
// Kernel 1: feat_sum over F (fp32 accumulate), store as f16 rows (B x 64).
// Carries the dominant 163.6 MB HBM read; fully coalesced streaming.
// ---------------------------------------------------------------------------
__global__ __launch_bounds__(256) void k_featsum(const float* __restrict__ embeds,
                                                 _Float16* __restrict__ feat_h) {
  const int idx = blockIdx.x * 256 + threadIdx.x;   // B*E threads
  const int e = idx & 63;
  const int b = idx >> 6;
  const float* p = embeds + (size_t)b * (F_TOT * E_TOT) + e;
  float s = 0.f;
  #pragma unroll
  for (int f = 0; f < F_TOT; ++f) s += p[(size_t)f * E_TOT];
  feat_h[idx] = (_Float16)s;
}

// ---------------------------------------------------------------------------
// Kernel 2: repack W (f32 -> f16) into WMMA B-fragment layout.
// Fragment (nt,kc) is a contiguous 1KB block; lane L owns 16 halves:
//   lane<16 : N=L,    K = kc*32 + 0..15
//   lane>=16: N=L-16, K = kc*32 + 16..31
// ---------------------------------------------------------------------------
__global__ __launch_bounds__(256) void k_packw(const float* __restrict__ w,
                                               _Float16* __restrict__ wp) {
  const int idx = blockIdx.x * 256 + threadIdx.x;   // NT*KCN*32*16 = 512K
  const int h  = idx & 15;
  const int L  = (idx >> 4) & 31;
  const int kc = (idx >> 9) & (KCN - 1);
  const int nt = idx >> 16;
  const int u  = nt * 16 + (L & 15);
  const int k  = kc * 32 + (L >> 4) * 16 + h;       // k = i*64 + j
  wp[idx] = (_Float16)w[(size_t)u * K_TOT + k];
}

// ---------------------------------------------------------------------------
// Kernel 3: lp = G @ W^T as WMMA f16 GEMM; G generated on the fly:
//   A-fragment(kc) = jvec[kc&1] * f[b, kc>>1]   (8x v_pk_mul_f16 / fragment)
// W fragments staged into LDS by the TDM (double buffered), shared by all
// 4 waves -> L2 traffic / 4. nt-range split over gridDim.y.
// ---------------------------------------------------------------------------
__global__ __launch_bounds__(WAVES * 32) void k_gemm(const _Float16* __restrict__ feat_h,
                                                     const _Float16* __restrict__ wp,
                                                     float* __restrict__ out) {
  __shared__ __align__(16) _Float16 wbuf[2][KB * 512];   // 2 x 16KB
  __shared__ _Float16 sF[WAVES][WT][16 * 64];            // 16KB
  const int lane = threadIdx.x & 31;
  const int w    = threadIdx.x >> 5;
  const int task = blockIdx.x * WAVES + w;
  const int ntBase = blockIdx.y * NT_PER;

  // Stage this wave's WT row-tiles of f (2KB each, contiguous) into LDS for
  // the per-kc scalar broadcast f[b, kc>>1].
  for (int t = 0; t < WT; ++t) {
    const int b0 = (task * WT + t) * 16;
    const uint4* src = (const uint4*)(feat_h + (size_t)b0 * 64);
    uint4* dst = (uint4*)&sF[w][t][0];
    #pragma unroll
    for (int r = 0; r < 4; ++r) dst[lane + 32 * r] = src[lane + 32 * r];
  }

  // Preload per-lane j-vectors (parity 0: j in [0,32), parity 1: j in [32,64)).
  V16 jv[WT][2];
  #pragma unroll
  for (int t = 0; t < WT; ++t) {
    const int b = (task * WT + t) * 16 + (lane & 15);
    const int g = (lane >> 4) * 16;
    #pragma unroll
    for (int par = 0; par < 2; ++par) {
      const _Float16* base = feat_h + (size_t)b * 64 + par * 32 + g;
      jv[t][par].h[0] = *(const v8h*)(base);        // K = 0..7   (+16 for hi lanes)
      jv[t][par].h[1] = *(const v8h*)(base + 8);    // K = 8..15  (+16 for hi lanes)
    }
  }

  // Prologue: wave 0 issues TDM load of window 0 into buffer 0.
  if (threadIdx.x < 32) {
    tdm_load_1d((unsigned)(uintptr_t)&wbuf[0][0],
                wp + (size_t)ntBase * KCN * 512, (KB * 512 * 2) / 8);
  }

  const int row = (lane & 15) * 64;
  const v8f zero = {};
  v8f acc[WT];

  for (int widx = 0; widx < NWIN; ++widx) {
    const int cur = widx & 1;
    const int nt  = ntBase + widx / (KCN / KB);
    const int kt  = widx % (KCN / KB);

    if (threadIdx.x < 32) {
      if (widx + 1 < NWIN) {
        const int nnt = ntBase + (widx + 1) / (KCN / KB);
        const int nkt = (widx + 1) % (KCN / KB);
        tdm_load_1d((unsigned)(uintptr_t)&wbuf[cur ^ 1][0],
                    wp + (size_t)(nnt * KCN + nkt * KB) * 512, (KB * 512 * 2) / 8);
        __builtin_amdgcn_s_wait_tensorcnt(1);   // window widx complete
      } else {
        __builtin_amdgcn_s_wait_tensorcnt(0);
      }
    }
    __syncthreads();          // publish wbuf[cur] to all waves

    if (kt == 0) {
      #pragma unroll
      for (int t = 0; t < WT; ++t) acc[t] = zero;
    }

    const _Float16* wb = &wbuf[cur][lane * 16];
    #pragma unroll 4
    for (int kl = 0; kl < KB; ++kl) {
      const int kc  = kt * KB + kl;
      const v16h bf = *(const v16h*)(wb + (size_t)kl * 512);   // 2x ds_load_b128
      const int i   = kc >> 1;
      const int par = kc & 1;
      #pragma unroll
      for (int t = 0; t < WT; ++t) {
        const _Float16 s = sF[w][t][row + i];
        const v16h a = jv[t][par].v * s;                       // 8x v_pk_mul_f16
        acc[t] = __builtin_amdgcn_wmma_f32_16x16x32_f16(
            /*neg_a=*/false, a, /*neg_b=*/false, bf,
            /*c_mod=*/(short)0, acc[t], /*reuse_a=*/false, /*reuse_b=*/false);
      }
    }

    if (kt == (KCN / KB) - 1) {
      // D layout: lane L, VGPR v -> (M = v + 8*(L>=16), N = L&15)
      #pragma unroll
      for (int t = 0; t < WT; ++t) {
        const int b0 = (task * WT + t) * 16 + 8 * (lane >> 4);
        const int u  = nt * 16 + (lane & 15);
        #pragma unroll
        for (int v = 0; v < 8; ++v)
          out[(size_t)(b0 + v) * U_TOT + u] = acc[t][v];
      }
    }
    __syncthreads();          // all waves done with wbuf[cur] before reuse
  }
}

// ---------------------------------------------------------------------------
extern "C" void kernel_launch(void* const* d_in, const int* in_sizes, int n_in,
                              void* d_out, int out_size, void* d_ws, size_t ws_size,
                              hipStream_t stream) {
  const float* embeds  = (const float*)d_in[0];  // (B, F, E) fp32
  const float* outer_w = (const float*)d_in[1];  // (U, E, E) fp32
  float* out = (float*)d_out;                    // (B, U) fp32

  _Float16* feat_h = (_Float16*)d_ws;                       // 2 MB
  _Float16* wpack  = feat_h + (size_t)B_TOT * E_TOT;        // 1 MB

  k_featsum<<<(B_TOT * E_TOT) / 256, 256, 0, stream>>>(embeds, feat_h);
  k_packw<<<(NT * KCN * 32 * 16) / 256, 256, 0, stream>>>(outer_w, wpack);

  dim3 grid((B_TOT / 16) / (WAVES * WT), GRID_Y);   // 128 x 2 blocks, 4 waves
  k_gemm<<<grid, WAVES * 32, 0, stream>>>(feat_h, wpack, out);
}
